// Attention_regular_24464133718790
// MI455X (gfx1250) — matrix operand order repaired and checked by reading, not compiled, not run on hardware
//
#include <hip/hip_runtime.h>
#include <hip/hip_bf16.h>

typedef _Float16 h8  __attribute__((ext_vector_type(8)));
typedef _Float16 v16h __attribute__((ext_vector_type(16)));
typedef float    v8f  __attribute__((ext_vector_type(8)));

#define NUM_HEADS 6
#define HEAD_DIM  32
#define H_SP      8
#define W_SP      32
#define N_TOK     256          // tokens per window
#define IMG       256          // H == W == 256
#define CH        192          // DIM
#define NW_H      32           // IMG / H_SP
#define NW_W      8            // IMG / W_SP
#define NUM_BIAS  945          // (2*8-1)*(2*32-1)
#define QKV_STRIDE 25165824LL  // B*L*C = 2*65536*192
#define SCALE_F   0.17677669529663687f  // 1/sqrt(32)

static __device__ __forceinline__ v16h cat16(h8 a, h8 b) {
    return __builtin_shufflevector(a, b, 0,1,2,3,4,5,6,7,8,9,10,11,12,13,14,15);
}

// ---------------------------------------------------------------------------
// Kernel 1: DynamicPosBias MLP  (945 rows, scalar per thread — negligible cost)
// ---------------------------------------------------------------------------
__global__ void pos_mlp_kernel(const float* __restrict__ biases,
                               const float* __restrict__ pw,  const float* __restrict__ pb,
                               const float* __restrict__ g1,  const float* __restrict__ b1,
                               const float* __restrict__ w1,  const float* __restrict__ fb1,
                               const float* __restrict__ g2,  const float* __restrict__ b2,
                               const float* __restrict__ w2,  const float* __restrict__ fb2,
                               const float* __restrict__ g3,  const float* __restrict__ b3,
                               const float* __restrict__ w3,  const float* __restrict__ fb3,
                               float* __restrict__ pos) {
    int i = blockIdx.x * blockDim.x + threadIdx.x;
    if (i >= NUM_BIAS) return;
    float x[12], y[12];
    float in0 = biases[i * 2 + 0], in1 = biases[i * 2 + 1];
    #pragma unroll
    for (int o = 0; o < 12; ++o) x[o] = in0 * pw[o] + in1 * pw[12 + o] + pb[o];

    const float* G[3]  = {g1, g2, g3};
    const float* Bb[3] = {b1, b2, b3};
    #pragma unroll
    for (int blk = 0; blk < 3; ++blk) {
        float m = 0.f;
        #pragma unroll
        for (int j = 0; j < 12; ++j) m += x[j];
        m *= (1.f / 12.f);
        float v = 0.f;
        #pragma unroll
        for (int j = 0; j < 12; ++j) { float d = x[j] - m; v += d * d; }
        v *= (1.f / 12.f);
        float rs = rsqrtf(v + 1e-5f);
        #pragma unroll
        for (int j = 0; j < 12; ++j) {
            float t = (x[j] - m) * rs * G[blk][j] + Bb[blk][j];
            x[j] = t > 0.f ? t : 0.f;
        }
        if (blk == 0) {
            #pragma unroll
            for (int o = 0; o < 12; ++o) { float s = fb1[o];
                #pragma unroll
                for (int j = 0; j < 12; ++j) s += x[j] * w1[j * 12 + o];
                y[o] = s; }
            #pragma unroll
            for (int o = 0; o < 12; ++o) x[o] = y[o];
        } else if (blk == 1) {
            #pragma unroll
            for (int o = 0; o < 12; ++o) { float s = fb2[o];
                #pragma unroll
                for (int j = 0; j < 12; ++j) s += x[j] * w2[j * 12 + o];
                y[o] = s; }
            #pragma unroll
            for (int o = 0; o < 12; ++o) x[o] = y[o];
        }
    }
    #pragma unroll
    for (int o = 0; o < NUM_HEADS; ++o) {
        float s = fb3[o];
        #pragma unroll
        for (int j = 0; j < 12; ++j) s += x[j] * w3[j * NUM_HEADS + o];
        pos[i * NUM_HEADS + o] = s;
    }
}

// ---------------------------------------------------------------------------
// Kernel 2: expand rel bias directly into WMMA C-fragment layout:
//   relBias[((h*256 + qt*16+kt)*32 + lane)*8 + r]
//     = pos[relIndex[(qt*16 + r + 8*(lane>>4))*256 + kt*16 + (lane&15)]][h]
// so each lane of the attention kernel reads its 8 C values contiguously.
// ---------------------------------------------------------------------------
__global__ void bias_expand_kernel(const float* __restrict__ pos,
                                   const int* __restrict__ relIndex,
                                   float* __restrict__ relBias) {
    int idx = blockIdx.x * blockDim.x + threadIdx.x;   // 6*65536 total
    int h    = idx >> 16;
    int rem  = idx & 65535;
    int t    = rem >> 8;          // qt*16 + kt
    int lane = (rem >> 3) & 31;
    int r    = idx & 7;
    int qt = t >> 4, kt = t & 15;
    int hi = lane >> 4, lo = lane & 15;
    int q = qt * 16 + r + 8 * hi;
    int k = kt * 16 + lo;
    relBias[idx] = pos[relIndex[q * N_TOK + k] * NUM_HEADS + h];
}

// ---------------------------------------------------------------------------
// Kernel 3: windowed attention. 1 block = 1 (window, head). 256 thr = 8 waves.
// ---------------------------------------------------------------------------
__global__ __launch_bounds__(256) void attn_kernel(const float* __restrict__ qkv,
                                                   const float* __restrict__ relBias,
                                                   float* __restrict__ out) {
    __shared__ _Float16 Klds[N_TOK][40];      // K, f16, row-major      (20480 B)
    __shared__ _Float16 Vt[HEAD_DIM][264];    // V transposed, f16      (16896 B)
    __shared__ _Float16 Pst[8][2][16][40];    // per-wave 2x P staging  (20480 B)

    const int tid  = threadIdx.x;
    const int wave = tid >> 5;
    const int lane = tid & 31;
    const int lo   = lane & 15;
    const int hi   = lane >> 4;

    int blk = blockIdx.x;
    const int head = blk % NUM_HEADS;  blk /= NUM_HEADS;
    const int wB   = blk % NW_W;       blk /= NW_W;
    const int hB   = blk % NW_H;       blk /= NW_H;
    const int b    = blk;

    // element offset of (token 0, dim 0) of this window/head inside one qkv slab
    const long long winBase =
        (((long long)b * IMG + hB * H_SP) * IMG + wB * W_SP) * CH + head * HEAD_DIM;

    // ---- cooperative load: K -> Klds (f16), V -> Vt (f16 transposed) ----
    {
        const int n = tid;  // token
        const long long off = winBase + (long long)(((n >> 5) * IMG) + (n & 31)) * CH;
        const float4* kp = (const float4*)(qkv + QKV_STRIDE + off);
        const float4* vp = (const float4*)(qkv + 2 * QKV_STRIDE + off);
        #pragma unroll
        for (int i = 0; i < 8; ++i) {
            float4 kv = kp[i];
            Klds[n][i * 4 + 0] = (_Float16)kv.x;
            Klds[n][i * 4 + 1] = (_Float16)kv.y;
            Klds[n][i * 4 + 2] = (_Float16)kv.z;
            Klds[n][i * 4 + 3] = (_Float16)kv.w;
            float4 vv = vp[i];
            Vt[i * 4 + 0][n] = (_Float16)vv.x;
            Vt[i * 4 + 1][n] = (_Float16)vv.y;
            Vt[i * 4 + 2][n] = (_Float16)vv.z;
            Vt[i * 4 + 3][n] = (_Float16)vv.w;
        }
    }
    __syncthreads();

    // bias fragments for this head, laid out [tile][lane][8]
    const float* biasH = relBias + head * (N_TOK * N_TOK);

    for (int s = 0; s < 2; ++s) {
        const int q0 = wave * 32 + s * 16;   // q-strip base row
        const int qt = q0 >> 4;              // q tile index

        // ---- Q A-fragment straight from global (fold in softmax scale) ----
        const int qrow = q0 + lo;
        const long long qoff =
            winBase + (long long)(((qrow >> 5) * IMG) + (qrow & 31)) * CH;
        const float4* qp = (const float4*)(qkv + qoff);
        float4 f0 = qp[hi * 2 + 0];
        float4 f1 = qp[hi * 2 + 1];
        float4 f2 = qp[4 + hi * 2 + 0];
        float4 f3 = qp[4 + hi * 2 + 1];
        v16h aQ;
        aQ[0]  = (_Float16)(f0.x * SCALE_F); aQ[1]  = (_Float16)(f0.y * SCALE_F);
        aQ[2]  = (_Float16)(f0.z * SCALE_F); aQ[3]  = (_Float16)(f0.w * SCALE_F);
        aQ[4]  = (_Float16)(f1.x * SCALE_F); aQ[5]  = (_Float16)(f1.y * SCALE_F);
        aQ[6]  = (_Float16)(f1.z * SCALE_F); aQ[7]  = (_Float16)(f1.w * SCALE_F);
        aQ[8]  = (_Float16)(f2.x * SCALE_F); aQ[9]  = (_Float16)(f2.y * SCALE_F);
        aQ[10] = (_Float16)(f2.z * SCALE_F); aQ[11] = (_Float16)(f2.w * SCALE_F);
        aQ[12] = (_Float16)(f3.x * SCALE_F); aQ[13] = (_Float16)(f3.y * SCALE_F);
        aQ[14] = (_Float16)(f3.z * SCALE_F); aQ[15] = (_Float16)(f3.w * SCALE_F);

        // ---- S = Q*Kt + bias : 16 WMMAs, bias fragment preloaded as C ----
        v8f acc[16];
        #pragma unroll
        for (int t = 0; t < 16; ++t) {
            const h8* kr = (const h8*)&Klds[t * 16 + lo][hi * 8];
            v16h bK = cat16(kr[0], kr[2]);
            v8f c = *(const v8f*)(biasH + (((qt << 4) + t) << 8) + lane * 8);
            acc[t] = __builtin_amdgcn_wmma_f32_16x16x32_f16(
                false, aQ, false, bK, (short)0, c, false, false);
        }

        // ---- softmax over k (rows fixed per (vgpr, half-wave) slot) ----
        v8f rinv;
        #pragma unroll
        for (int r = 0; r < 8; ++r) {
            float m = acc[0][r];
            #pragma unroll
            for (int t = 1; t < 16; ++t) m = fmaxf(m, acc[t][r]);
            m = fmaxf(m, __shfl_xor(m, 1));
            m = fmaxf(m, __shfl_xor(m, 2));
            m = fmaxf(m, __shfl_xor(m, 4));
            m = fmaxf(m, __shfl_xor(m, 8));
            float sum = 0.f;
            #pragma unroll
            for (int t = 0; t < 16; ++t) {
                float e = __expf(acc[t][r] - m);
                acc[t][r] = e;       // keep unnormalized; fold 1/sum into output
                sum += e;
            }
            sum += __shfl_xor(sum, 1);
            sum += __shfl_xor(sum, 2);
            sum += __shfl_xor(sum, 4);
            sum += __shfl_xor(sum, 8);
            rinv[r] = __builtin_amdgcn_rcpf(sum);
        }

        // ---- O = P~*V with double-buffered 16x32 P staging tiles ----
        v8f o0 = {}, o1 = {};
        // prologue: stage chunk 0 into buffer 0
        #pragma unroll
        for (int u = 0; u < 2; ++u)
            #pragma unroll
            for (int r = 0; r < 8; ++r)
                Pst[wave][0][r + 8 * hi][u * 16 + lo] = (_Float16)acc[u][r];

        #pragma unroll
        for (int kc = 0; kc < 8; ++kc) {
            const int cur = kc & 1;
            if (kc < 7) {   // stage next chunk first; its stores overlap our WMMAs
                const int nxt = (kc + 1) & 1;
                #pragma unroll
                for (int u = 0; u < 2; ++u)
                    #pragma unroll
                    for (int r = 0; r < 8; ++r)
                        Pst[wave][nxt][r + 8 * hi][u * 16 + lo] =
                            (_Float16)acc[2 * (kc + 1) + u][r];
            }
            v16h aP = cat16(*(const h8*)&Pst[wave][cur][lo][hi * 8],
                            *(const h8*)&Pst[wave][cur][lo][16 + hi * 8]);
            const h8* v0 = (const h8*)&Vt[lo][kc * 32 + hi * 8];
            const h8* v1 = (const h8*)&Vt[16 + lo][kc * 32 + hi * 8];
            v16h bV0 = cat16(v0[0], v0[2]);
            v16h bV1 = cat16(v1[0], v1[2]);
            o0 = __builtin_amdgcn_wmma_f32_16x16x32_f16(
                false, aP, false, bV0, (short)0, o0, false, false);
            o1 = __builtin_amdgcn_wmma_f32_16x16x32_f16(
                false, aP, false, bV1, (short)0, o1, false, false);
        }

        // ---- store output (B, H, W, C); apply softmax 1/sum per row ----
        #pragma unroll
        for (int r = 0; r < 8; ++r) {
            int row = q0 + r + 8 * hi;
            long long ooff =
                winBase + (long long)(((row >> 5) * IMG) + (row & 31)) * CH;
            out[ooff + lo]      = o0[r] * rinv[r];
            out[ooff + 16 + lo] = o1[r] * rinv[r];
        }
    }
}

// ---------------------------------------------------------------------------
extern "C" void kernel_launch(void* const* d_in, const int* in_sizes, int n_in,
                              void* d_out, int out_size, void* d_ws, size_t ws_size,
                              hipStream_t stream) {
    (void)in_sizes; (void)n_in; (void)out_size; (void)ws_size;
    const float* qkv    = (const float*)d_in[0];
    const float* rpe    = (const float*)d_in[3];
    const int*   relIdx = (const int*)d_in[4];
    const float* pw  = (const float*)d_in[5],  *pb  = (const float*)d_in[6];
    const float* g1  = (const float*)d_in[7],  *b1  = (const float*)d_in[8];
    const float* w1  = (const float*)d_in[9],  *fb1 = (const float*)d_in[10];
    const float* g2  = (const float*)d_in[11], *b2  = (const float*)d_in[12];
    const float* w2  = (const float*)d_in[13], *fb2 = (const float*)d_in[14];
    const float* g3  = (const float*)d_in[15], *b3  = (const float*)d_in[16];
    const float* w3  = (const float*)d_in[17], *fb3 = (const float*)d_in[18];

    float* pos     = (float*)d_ws;                         // 945*6 f32
    float* relBias = (float*)((char*)d_ws + 32768);        // 6*65536 f32, fragment layout
    float* outp    = (float*)d_out;

    pos_mlp_kernel<<<4, 256, 0, stream>>>(rpe, pw, pb, g1, b1, w1, fb1,
                                          g2, b2, w2, fb2, g3, b3, w3, fb3, pos);
    bias_expand_kernel<<<(NUM_HEADS * 65536) / 256, 256, 0, stream>>>(pos, relIdx, relBias);
    attn_kernel<<<2 * NW_H * NW_W * NUM_HEADS, 256, 0, stream>>>(qkv, relBias, outp);
}
